// view_int_24567212933534
// MI455X (gfx1250) — matrix operand order repaired and checked by reading, not compiled, not run on hardware
//
#include <hip/hip_runtime.h>
#include <hip/hip_bf16.h>

#define HD 256      // hidden dim
#define K2 512      // 2*H contraction for graph linears

typedef __attribute__((ext_vector_type(16))) __bf16 v16bf;
typedef __attribute__((ext_vector_type(8)))  float  v8f;
typedef __attribute__((ext_vector_type(4)))  float  v4f;

// ---------------------------------------------------------------------------
// A-fragment: 16x32 bf16 from f32 row-major source (optionally split xa|xb).
// lane<16: row=lane,  K = {0..15}
// lane>=16: row=lane-16, K = {16..31}
// ---------------------------------------------------------------------------
__device__ inline v16bf load_a_frag(const float* __restrict__ rowa,
                                    const float* __restrict__ rowb,
                                    int k0, int akb) {
  const float* src = (rowb != nullptr && k0 >= HD) ? (rowb + (k0 - HD)) : (rowa + k0);
  const float* p = src + akb;              // 32B-aligned
  v4f c0 = *(const v4f*)(p);
  v4f c1 = *(const v4f*)(p + 4);
  v4f c2 = *(const v4f*)(p + 8);
  v4f c3 = *(const v4f*)(p + 12);
  v16bf f;
#pragma unroll
  for (int i = 0; i < 4; ++i) {
    f[i]      = (__bf16)c0[i];
    f[4 + i]  = (__bf16)c1[i];
    f[8 + i]  = (__bf16)c2[i];
    f[12 + i] = (__bf16)c3[i];
  }
  return f;
}

__device__ inline void store_tile_tanh(float* __restrict__ out, const v8f& acc,
                                       const float* __restrict__ bias,
                                       int mbase, int col) {
  float b = bias[col];
#pragma unroll
  for (int v = 0; v < 8; ++v)
    out[(size_t)(mbase + v) * HD + col] = tanhf(acc[v] + b);
}

__device__ inline float score_tile(const v8f& acc, const float* __restrict__ b1,
                                   const float* __restrict__ w2, int col) {
  float b = b1[col], w = w2[col], p = 0.f;
#pragma unroll
  for (int v = 0; v < 8; ++v) p += tanhf(acc[v] + b) * w;
  return p;
}

// ---------------------------------------------------------------------------
// Y[R x 256] = tanh( cat(xa,xb)[rows] @ W + b ),  W supplied transposed bf16
// wT: [HD][K] bf16, row stride K. ridx: optional row gather (movie_id).
// Wave tile: 16(M) x 64(N); 8 waves/block.
// ---------------------------------------------------------------------------
__global__ __launch_bounds__(256)
void gemm_tanh_kernel(const float* __restrict__ xa, const float* __restrict__ xb,
                      const int* __restrict__ ridx,
                      const __bf16* __restrict__ wT, const int K,
                      const float* __restrict__ bias,
                      float* __restrict__ out, const int R) {
  const int lane = threadIdx.x & 31;
  const int wave = threadIdx.x >> 5;
  const int tile = blockIdx.x * 8 + wave;
  const int m0 = (tile >> 2) * 16;       // HD/64 == 4 n-groups
  const int n0 = (tile & 3) * 64;
  if (m0 >= R) return;

  const int ar   = m0 + (lane & 15);
  const int rowg = ridx ? ridx[ar] : ar;
  const float* rowa = xa + (size_t)rowg * HD;
  const float* rowb = xb ? (xb + (size_t)rowg * HD) : nullptr;
  const int akb = (lane < 16) ? 0 : 16;
  const int cn  = lane & 15;
  const int bkb = (lane < 16) ? 0 : 16;  // B: lane<16 K=0..15, lane>=16 K=16..31

  v8f acc0 = {}, acc1 = {}, acc2 = {}, acc3 = {};
  for (int k0 = 0; k0 < K; k0 += 32) {
    v16bf a = load_a_frag(rowa, rowb, k0, akb);
    const __bf16* bp = wT + (size_t)(n0 + cn) * K + k0 + bkb;
    v16bf b0 = *(const v16bf*)(bp);
    v16bf b1 = *(const v16bf*)(bp + 16 * K);
    v16bf b2 = *(const v16bf*)(bp + 32 * K);
    v16bf b3 = *(const v16bf*)(bp + 48 * K);
    acc0 = __builtin_amdgcn_wmma_f32_16x16x32_bf16(false, a, false, b0, (short)0, acc0, false, false);
    acc1 = __builtin_amdgcn_wmma_f32_16x16x32_bf16(false, a, false, b1, (short)0, acc1, false, false);
    acc2 = __builtin_amdgcn_wmma_f32_16x16x32_bf16(false, a, false, b2, (short)0, acc2, false, false);
    acc3 = __builtin_amdgcn_wmma_f32_16x16x32_bf16(false, a, false, b3, (short)0, acc3, false, false);
  }
  const int mbase = m0 + ((lane < 16) ? 0 : 8);
  store_tile_tanh(out, acc0, bias, mbase, n0 + cn);
  store_tile_tanh(out, acc1, bias, mbase, n0 + 16 + cn);
  store_tile_tanh(out, acc2, bias, mbase, n0 + 32 + cn);
  store_tile_tanh(out, acc3, bias, mbase, n0 + 48 + cn);
}

// ---------------------------------------------------------------------------
// score += sum_{r,h} tanh( (s @ W1 + b1) )_{r,h} * W2_h      (K = HD)
// ---------------------------------------------------------------------------
__global__ __launch_bounds__(256)
void score_kernel(const float* __restrict__ s,
                  const __bf16* __restrict__ w1T,
                  const float* __restrict__ b1,
                  const float* __restrict__ w2,
                  float* __restrict__ score, const int R) {
  const int lane = threadIdx.x & 31;
  const int wave = threadIdx.x >> 5;
  const int tile = blockIdx.x * 8 + wave;
  const int m0 = (tile >> 2) * 16;
  const int n0 = (tile & 3) * 64;
  if (m0 >= R) return;

  const float* rowa = s + (size_t)(m0 + (lane & 15)) * HD;
  const int akb = (lane < 16) ? 0 : 16;
  const int cn  = lane & 15;
  const int bkb = (lane < 16) ? 0 : 16;

  v8f acc0 = {}, acc1 = {}, acc2 = {}, acc3 = {};
  for (int k0 = 0; k0 < HD; k0 += 32) {
    v16bf a = load_a_frag(rowa, nullptr, k0, akb);
    const __bf16* bp = w1T + (size_t)(n0 + cn) * HD + k0 + bkb;
    v16bf b0 = *(const v16bf*)(bp);
    v16bf b1 = *(const v16bf*)(bp + 16 * HD);
    v16bf b2 = *(const v16bf*)(bp + 32 * HD);
    v16bf b3 = *(const v16bf*)(bp + 48 * HD);
    acc0 = __builtin_amdgcn_wmma_f32_16x16x32_bf16(false, a, false, b0, (short)0, acc0, false, false);
    acc1 = __builtin_amdgcn_wmma_f32_16x16x32_bf16(false, a, false, b1, (short)0, acc1, false, false);
    acc2 = __builtin_amdgcn_wmma_f32_16x16x32_bf16(false, a, false, b2, (short)0, acc2, false, false);
    acc3 = __builtin_amdgcn_wmma_f32_16x16x32_bf16(false, a, false, b3, (short)0, acc3, false, false);
  }
  float part = 0.f;
  part += score_tile(acc0, b1, w2, n0 + cn);
  part += score_tile(acc1, b1, w2, n0 + 16 + cn);
  part += score_tile(acc2, b1, w2, n0 + 32 + cn);
  part += score_tile(acc3, b1, w2, n0 + 48 + cn);
#pragma unroll
  for (int o = 16; o > 0; o >>= 1) part += __shfl_xor(part, o, 32);
  if (lane == 0) atomicAdd(score, part);
}

// --------------------------- small helper kernels ---------------------------
__global__ void transcvt_kernel(const float* __restrict__ W, __bf16* __restrict__ WT,
                                const int K) {  // W: [K][HD] f32 -> WT: [HD][K] bf16
  int idx = blockIdx.x * blockDim.x + threadIdx.x;
  if (idx >= K * HD) return;
  int n = idx / K, k = idx - n * K;
  WT[idx] = (__bf16)W[(size_t)k * HD + n];
}

__global__ void init_kernel(int* __restrict__ map, float* __restrict__ scores, const int Nn) {
  int i = blockIdx.x * blockDim.x + threadIdx.x;
  if (i < Nn) map[i] = -1;
  if (i < 4) scores[i] = 0.f;
}

__global__ void scatter_kernel(const int* __restrict__ mid, int* __restrict__ map, const int Mm) {
  int j = blockIdx.x * blockDim.x + threadIdx.x;
  if (j < Mm) map[mid[j]] = j;
}

__global__ void softmax_kernel(const float* __restrict__ scores,
                               const float* b2m1, const float* b2m2,
                               const float* b2r1, const float* b2r2,
                               const float invM, const float invB,
                               float* __restrict__ wts) {
  float s0 = b2m1[0] + scores[0] * invM;
  float s1 = b2m2[0] + scores[1] * invM;
  float s2 = b2r1[0] + scores[2] * invB;
  float s3 = b2r2[0] + scores[3] * invB;
  float m = fmaxf(s0, s1);
  float e0 = expf(s0 - m), e1 = expf(s1 - m);
  wts[0] = e0 / (e0 + e1); wts[1] = e1 / (e0 + e1);
  m = fmaxf(s2, s3);
  float e2 = expf(s2 - m), e3 = expf(s3 - m);
  wts[2] = e2 / (e2 + e3); wts[3] = e3 / (e2 + e3);
}

// out layout: 6 arrays of [N][HD] concatenated. Review replacement (rows <B,
// arrays 2..5) overrides movie replacement (mapped rows, arrays 0..3), matching
// the reference's .at[mid].set(...).at[:B].set(...) ordering.
__global__ __launch_bounds__(256)
void assemble_kernel(const float* __restrict__ in0, const float* __restrict__ in1,
                     const float* __restrict__ in2, const float* __restrict__ in3,
                     const float* __restrict__ in4, const float* __restrict__ in5,
                     const float* __restrict__ s1m, const float* __restrict__ s2m,
                     const float* __restrict__ s2r, const float* __restrict__ s3r,
                     const float* __restrict__ wts, const int* __restrict__ map,
                     const int Nn, const int Bb, float* __restrict__ out) {
  const size_t per4 = (size_t)Nn * (HD / 4);
  size_t tid = (size_t)blockIdx.x * blockDim.x + threadIdx.x;
  if (tid >= per4 * 6) return;
  int a = (int)(tid / per4);
  size_t rem = tid - (size_t)a * per4;
  int r = (int)(rem / (HD / 4));
  int c = (int)(rem - (size_t)r * (HD / 4)) * 4;
  int j = map[r];
  v4f val;
  if (r < Bb && a >= 2) {
    v4f x = *(const v4f*)(s2r + (size_t)r * HD + c);
    v4f y = *(const v4f*)(s3r + (size_t)r * HD + c);
    val = wts[2] * x + wts[3] * y;
  } else if (j >= 0 && a <= 3) {
    v4f x = *(const v4f*)(s1m + (size_t)j * HD + c);
    v4f y = *(const v4f*)(s2m + (size_t)j * HD + c);
    val = wts[0] * x + wts[1] * y;
  } else {
    const float* src = a == 0 ? in0 : a == 1 ? in1 : a == 2 ? in2
                      : a == 3 ? in3 : a == 4 ? in4 : in5;
    val = *(const v4f*)(src + (size_t)r * HD + c);
  }
  *((v4f*)out + tid) = val;
}

// ---------------------------------------------------------------------------
extern "C" void kernel_launch(void* const* d_in, const int* in_sizes, int n_in,
                              void* d_out, int out_size, void* d_ws, size_t ws_size,
                              hipStream_t stream) {
  const float* sub1_text   = (const float*)d_in[0];
  const float* sub1_struct = (const float*)d_in[1];
  const float* sub2_text   = (const float*)d_in[2];
  const float* sub2_meta   = (const float*)d_in[3];
  const float* sub3_text   = (const float*)d_in[4];
  const float* sub3_meta   = (const float*)d_in[5];
  const int*   movie_id    = (const int*)d_in[6];
  const float* g1_W = (const float*)d_in[9];  const float* g1_b = (const float*)d_in[10];
  const float* g2_W = (const float*)d_in[11]; const float* g2_b = (const float*)d_in[12];
  const float* g3_W = (const float*)d_in[13]; const float* g3_b = (const float*)d_in[14];
  const float* m1W1 = (const float*)d_in[15]; const float* m1b1 = (const float*)d_in[16];
  const float* m1W2 = (const float*)d_in[17]; const float* m1b2 = (const float*)d_in[18];
  const float* m2W1 = (const float*)d_in[19]; const float* m2b1 = (const float*)d_in[20];
  const float* m2W2 = (const float*)d_in[21]; const float* m2b2 = (const float*)d_in[22];
  const float* r1W1 = (const float*)d_in[23]; const float* r1b1 = (const float*)d_in[24];
  const float* r1W2 = (const float*)d_in[25]; const float* r1b2 = (const float*)d_in[26];
  const float* r2W1 = (const float*)d_in[27]; const float* r2b1 = (const float*)d_in[28];
  const float* r2W2 = (const float*)d_in[29]; const float* r2b2 = (const float*)d_in[30];

  const int M = in_sizes[6];        // 8192 movie rows
  const int N = in_sizes[0] / HD;   // 65536 nodes
  const int B = 16384;              // batch_size (reference constant)

  // workspace carve (all sub-buffers naturally 256B aligned)
  char* w = (char*)d_ws;
  float* s1m = (float*)w; w += (size_t)M * HD * 4;
  float* s2m = (float*)w; w += (size_t)M * HD * 4;
  float* s2r = (float*)w; w += (size_t)B * HD * 4;
  float* s3r = (float*)w; w += (size_t)B * HD * 4;
  __bf16* g1T = (__bf16*)w; w += (size_t)K2 * HD * 2;
  __bf16* g2T = (__bf16*)w; w += (size_t)K2 * HD * 2;
  __bf16* g3T = (__bf16*)w; w += (size_t)K2 * HD * 2;
  __bf16* a1T = (__bf16*)w; w += (size_t)HD * HD * 2;
  __bf16* a2T = (__bf16*)w; w += (size_t)HD * HD * 2;
  __bf16* a3T = (__bf16*)w; w += (size_t)HD * HD * 2;
  __bf16* a4T = (__bf16*)w; w += (size_t)HD * HD * 2;
  float* scores = (float*)w; w += 256;
  float* wts    = (float*)w; w += 256;
  int*   map    = (int*)w;   w += (size_t)N * 4;

  // 1) init per-launch state (harness does not re-zero ws between replays)
  init_kernel<<<(N + 255) / 256, 256, 0, stream>>>(map, scores, N);
  scatter_kernel<<<(M + 255) / 256, 256, 0, stream>>>(movie_id, map, M);

  // 2) transpose + bf16-convert the weight matrices
  transcvt_kernel<<<(K2 * HD + 255) / 256, 256, 0, stream>>>(g1_W, g1T, K2);
  transcvt_kernel<<<(K2 * HD + 255) / 256, 256, 0, stream>>>(g2_W, g2T, K2);
  transcvt_kernel<<<(K2 * HD + 255) / 256, 256, 0, stream>>>(g3_W, g3T, K2);
  transcvt_kernel<<<(HD * HD + 255) / 256, 256, 0, stream>>>(m1W1, a1T, HD);
  transcvt_kernel<<<(HD * HD + 255) / 256, 256, 0, stream>>>(m2W1, a2T, HD);
  transcvt_kernel<<<(HD * HD + 255) / 256, 256, 0, stream>>>(r1W1, a3T, HD);
  transcvt_kernel<<<(HD * HD + 255) / 256, 256, 0, stream>>>(r2W1, a4T, HD);

  // 3) graph linears (WMMA bf16, f32 accumulate)
  const int blkM = (M / 16) * (HD / 64) / 8;   // 256
  const int blkB = (B / 16) * (HD / 64) / 8;   // 512
  gemm_tanh_kernel<<<blkM, 256, 0, stream>>>(sub1_text, sub1_struct, movie_id, g1T, K2, g1_b, s1m, M);
  gemm_tanh_kernel<<<blkM, 256, 0, stream>>>(sub2_text, sub2_meta,   movie_id, g2T, K2, g2_b, s2m, M);
  gemm_tanh_kernel<<<blkB, 256, 0, stream>>>(sub2_text, sub2_meta,   nullptr,  g2T, K2, g2_b, s2r, B);
  gemm_tanh_kernel<<<blkB, 256, 0, stream>>>(sub3_text, sub3_meta,   nullptr,  g3T, K2, g3_b, s3r, B);

  // 4) attention scores (WMMA + wave reduce + atomic)
  score_kernel<<<blkM, 256, 0, stream>>>(s1m, a1T, m1b1, m1W2, scores + 0, M);
  score_kernel<<<blkM, 256, 0, stream>>>(s2m, a2T, m2b1, m2W2, scores + 1, M);
  score_kernel<<<blkB, 256, 0, stream>>>(s2r, a3T, r1b1, r1W2, scores + 2, B);
  score_kernel<<<blkB, 256, 0, stream>>>(s3r, a4T, r2b1, r2W2, scores + 3, B);

  // 5) softmax weights
  softmax_kernel<<<1, 1, 0, stream>>>(scores, m1b2, m2b2, r1b2, r2b2,
                                      1.0f / (float)M, 1.0f / (float)B, wts);

  // 6) fused copy + mix scatter into all six outputs
  size_t total4 = (size_t)6 * N * (HD / 4);
  assemble_kernel<<<(unsigned)((total4 + 255) / 256), 256, 0, stream>>>(
      sub1_text, sub1_struct, sub2_text, sub2_meta, sub3_text, sub3_meta,
      s1m, s2m, s2r, s3r, wts, map, N, B, (float*)d_out);
}
